// CrfRnn_36412732735943
// MI455X (gfx1250) — matrix operand order repaired and checked by reading, not compiled, not run on hardware
//
#include <hip/hip_runtime.h>

// CRF-RNN for MI455X (gfx1250).
// Spatial Gaussian: exact separable 1-D convolutions (cheap, f32).
// Bilateral Gaussian: dense N x N kernel generated ON THE FLY in the WMMA
// pipeline:  exponent tile = WMMA(featA_j, featB_i)  (features padded to K=32,
// with -0.5|f|^2 and 1.0 folded in as extra "feature" dims), then v_exp_f32,
// v_cvt_pk_rtz_f16 pack, used directly as the B operand (D layout = B's k order), and a second
// WMMA against q.  q row 31 is fixed at 1.0 so the same WMMA also produces the
// normalization row for free.  Working set (~6 MB) lives in the 192 MB L2.

typedef __attribute__((ext_vector_type(16))) _Float16 v16h;
typedef __attribute__((ext_vector_type(8)))  _Float16 v8h;
typedef __attribute__((ext_vector_type(2)))  __fp16   v2fp16;
typedef __attribute__((ext_vector_type(8)))  float    v8f;

#if __has_builtin(__builtin_amdgcn_exp2f)
#define EXP2F(x) __builtin_amdgcn_exp2f(x)
#else
#define EXP2F(x) exp2f(x)
#endif
#if __has_builtin(__builtin_amdgcn_rcpf)
#define RCPF(x) __builtin_amdgcn_rcpf(x)
#else
#define RCPF(x) (1.0f/(x))
#endif

__device__ __forceinline__ int pack2h(float a, float b) {
#if __has_builtin(__builtin_amdgcn_cvt_pkrtz)
  union { v2fp16 h; int d; } u;
  u.h = __builtin_amdgcn_cvt_pkrtz(a, b);
  return u.d;
#else
  union { int d; _Float16 h[2]; } u;
  u.h[0] = (_Float16)a; u.h[1] = (_Float16)b;
  return u.d;
#endif
}

static constexpr int   kN   = 32768;     // 32*32*32 points
static constexpr int   kL   = 21;        // labels
static constexpr int   kLp  = 32;        // padded labels
static constexpr float kAlpha = 160.0f;
static constexpr float kBeta  = 3.0f;
static constexpr float kGamma = 3.0f;
static constexpr float kLog2e = 1.44269504088896340736f;
static constexpr float kSqrtLog2e = 1.20112240878644980f;  // sqrt(log2 e)

// ---------------------------------------------------------------- setup: features
// featAp[n][16] = [f'0..f'5, h', 1.0,  0 x 8]   (A rows, j side; zero-padded)
// featBp[n][16] = [f'0..f'5, 1.0, h',  0 x 8]   (B cols, i side; zero-padded)
// WMMA dot over K gives  f'_i.f'_j + h'_j + h'_i = log2e * (-0.5 |f_i-f_j|^2).
// The zero pad lets hi lanes load their all-zero operand halves directly.
__global__ void k_feat(const float* __restrict__ img,
                       _Float16* __restrict__ featAp,
                       _Float16* __restrict__ featBp) {
  int n = blockIdx.x * blockDim.x + threadIdx.x;
  if (n >= kN) return;
  int z = n >> 10, y = (n >> 5) & 31, x = n & 31;
  float f[6];
  f[0] = ((float)z / kAlpha) * kSqrtLog2e;
  f[1] = ((float)y / kAlpha) * kSqrtLog2e;
  f[2] = ((float)x / kAlpha) * kSqrtLog2e;
  f[3] = (img[0 * kN + n] / kBeta) * kSqrtLog2e;
  f[4] = (img[1 * kN + n] / kBeta) * kSqrtLog2e;
  f[5] = (img[2 * kN + n] / kBeta) * kSqrtLog2e;
  float sq = 0.f;
#pragma unroll
  for (int k = 0; k < 6; ++k) sq += f[k] * f[k];
  float h = -0.5f * sq;
#pragma unroll
  for (int k = 0; k < 6; ++k) {
    featAp[(size_t)n * 16 + k] = (_Float16)f[k];
    featBp[(size_t)n * 16 + k] = (_Float16)f[k];
  }
  featAp[(size_t)n * 16 + 6] = (_Float16)h;
  featAp[(size_t)n * 16 + 7] = (_Float16)1.0f;
  featBp[(size_t)n * 16 + 6] = (_Float16)1.0f;
  featBp[(size_t)n * 16 + 7] = (_Float16)h;
#pragma unroll
  for (int k = 8; k < 16; ++k) {
    featAp[(size_t)n * 16 + k] = (_Float16)0.f;
    featBp[(size_t)n * 16 + k] = (_Float16)0.f;
  }
}

// ------------------------------------------- setup: spatial tables / q padding
__global__ void k_misc(float* __restrict__ gS, float* __restrict__ inv_ns,
                       _Float16* __restrict__ qh) {
  int n = blockIdx.x * blockDim.x + threadIdx.x;
  if (n >= kN) return;
  if (n < 32) {
    float t = (float)n / kGamma;
    gS[n] = EXP2F(-0.5f * kLog2e * t * t);
  }
  int z = n >> 10, y = (n >> 5) & 31, x = n & 31;
  float sz = 0.f, sy = 0.f, sx = 0.f;
  for (int k = 0; k < 32; ++k) {
    float dz = (float)(z - k) / kGamma, dy = (float)(y - k) / kGamma,
          dx = (float)(x - k) / kGamma;
    sz += EXP2F(-0.5f * kLog2e * dz * dz);
    sy += EXP2F(-0.5f * kLog2e * dy * dy);
    sx += EXP2F(-0.5f * kLog2e * dx * dx);
  }
  inv_ns[n] = RCPF(sz * sy * sx);
  // q pad rows: 21..30 = 0, row 31 = 1.0 (norm row: WMMA computes sum_j K[j,i])
#pragma unroll
  for (int r = kL; r < kLp - 1; ++r) qh[(size_t)r * kN + n] = (_Float16)0.f;
  qh[(size_t)(kLp - 1) * kN + n] = (_Float16)1.0f;
}

// ------------------------------------- setup: fold compatibility into weights
__global__ void k_foldw(const float* __restrict__ C,
                        const float* __restrict__ Ws,
                        const float* __restrict__ Wb,
                        float* __restrict__ cws, float* __restrict__ cwb) {
  int t = threadIdx.x;
  if (t >= kL * kL) return;
  const float* W = blockIdx.x ? Wb : Ws;
  float* o = blockIdx.x ? cwb : cws;
  int l = t / kL, m = t % kL;
  float a = 0.f;
  for (int k = 0; k < kL; ++k) a += C[l * kL + k] * W[k * kL + m];
  o[t] = a;
}

// --------------------------------------------------------------------- softmax
__global__ void k_softmax(const float* __restrict__ cur,
                          float* __restrict__ qf, _Float16* __restrict__ qh) {
  int n = blockIdx.x * blockDim.x + threadIdx.x;
  if (n >= kN) return;
  float v[kL];
  float mx = -3.4e38f;
#pragma unroll
  for (int l = 0; l < kL; ++l) { v[l] = cur[(size_t)l * kN + n]; mx = fmaxf(mx, v[l]); }
  float s = 0.f;
#pragma unroll
  for (int l = 0; l < kL; ++l) { float e = EXP2F((v[l] - mx) * kLog2e); v[l] = e; s += e; }
  float r = RCPF(s);
#pragma unroll
  for (int l = 0; l < kL; ++l) {
    float q = v[l] * r;
    qf[(size_t)l * kN + n] = q;
    if (qh) qh[(size_t)l * kN + n] = (_Float16)q;
  }
}

// --------------------------------------------- separable 1-D spatial gaussian
// STRIDE is compile-time so all 32 taps fold into immediate-offset loads.
template <int STRIDE>
__global__ void k_conv1d(const float* __restrict__ in, float* __restrict__ out,
                         const float* __restrict__ gS) {
  __shared__ float g[32];
  if (threadIdx.x < 32) g[threadIdx.x] = gS[threadIdx.x];
  __syncthreads();
  int t = blockIdx.x * blockDim.x + threadIdx.x;
  if (t >= kL * kN) return;
  int l = t / kN;
  int m = t - l * kN;
  int c = (m / STRIDE) & 31;
  int base = m - c * STRIDE;
  const float* row = in + (size_t)l * kN + base;
  float acc = 0.f;
#pragma unroll
  for (int k = 0; k < 32; ++k) {
    int d = c - k; d = d < 0 ? -d : d;
    acc += g[d] * row[k * STRIDE];
  }
  out[t] = acc;
}

// --------------------------------------------------- fused bilateral (4x WMMA)
__global__ __launch_bounds__(128)
void k_bilateral(const _Float16* __restrict__ featAp,
                 const _Float16* __restrict__ featBp,
                 const _Float16* __restrict__ qh,
                 float* __restrict__ bl, float* __restrict__ norm_b) {
  const int lane = threadIdx.x & 31;
  const int warp = threadIdx.x >> 5;
  const int i0   = (blockIdx.x * 4 + warp) * 16;   // this wave's 16-wide i tile
  const int half = lane >> 4;
  const int col  = lane & 15;
  const bool lo  = (half == 0);

  union U16 { v16h v; v8h h[2]; int d[8]; };

  // B operand of the exponent-WMMA: features of the 16 i-points (fixed).
  // 16-bit B(32x16): lanes 0-15 hold K=0..7,16..23, lanes 16-31 hold K=8..15,24..31.
  // Hi lanes read the zero pad of the same 32B row -> no masking VALU needed.
  U16 Bg;
  {
    const _Float16* pb = featBp + (size_t)(i0 + col) * 16;
    Bg.h[0] = *(const v8h*)(pb + (lo ? 0 : 8));
    Bg.h[1] = *(const v8h*)(pb + 8);
  }

  // Loop-invariant per-lane stream bases.
  const _Float16* pa  = featAp + (size_t)col * 16 + (lo ? 0 : 8);  // features/zeros
  const _Float16* paz = featAp + (size_t)col * 16 + 8;             // always zeros
  const _Float16* q0p = qh + (size_t)col * kN + half * 8;          // l = col
  const _Float16* q1p = qh + (size_t)(16 + col) * kN + half * 8;   // l = 16+col

  v8f c0 = {}; v8f c1 = {};   // D = [l x i] accumulators for l-tiles 0 and 1
  v8f zc = {};

#pragma unroll 2
  for (int jb = 0; jb < kN; jb += 32) {
    __builtin_prefetch(pa + (size_t)(jb + 64) * 16, 0, 3);
    __builtin_prefetch(q0p + jb + 64, 0, 3);
    // --- stage 1: exponent tiles  gram[j,i] = f'_j.f'_i + h'_j + h'_i
    U16 A0, A1;
    A0.h[0] = *(const v8h*)(pa  + (size_t)jb * 16);
    A0.h[1] = *(const v8h*)(paz + (size_t)jb * 16);
    A1.h[0] = *(const v8h*)(pa  + (size_t)(jb + 16) * 16);
    A1.h[1] = *(const v8h*)(paz + (size_t)(jb + 16) * 16);
    v8f g0 = __builtin_amdgcn_wmma_f32_16x16x32_f16(false, A0.v, false, Bg.v,
                                                    (short)0, zc, false, false);
    v8f g1 = __builtin_amdgcn_wmma_f32_16x16x32_f16(false, A1.v, false, Bg.v,
                                                    (short)0, zc, false, false);
    // --- stage 2: K = exp2(gram) (log2e pre-folded), pack straight to f16 pairs
    int P0[4], P1[4];
#pragma unroll
    for (int r = 0; r < 4; ++r) {
      P0[r] = pack2h(EXP2F(g0[2 * r]), EXP2F(g0[2 * r + 1]));
      P1[r] = pack2h(EXP2F(g1[2 * r]), EXP2F(g1[2 * r + 1]));
    }
    // D-layout (j rows split 0-7/8-15 across lane halves) already is the
    // B-operand table order (K=8h+s, 16+8h+s-8); the xor-16 shuffles are unused.
    int s0[4], s1[4];
#pragma unroll
    for (int r = 0; r < 4; ++r) {
      s0[r] = __shfl_xor(P0[r], 16, 32);
      s1[r] = __shfl_xor(P1[r], 16, 32);
    }
    U16 B2;
#pragma unroll
    for (int r = 0; r < 4; ++r) {
      B2.d[r]     = P0[r];   // K 0..7  (lo) / 8..15 (hi)
      B2.d[4 + r] = P1[r];   // K 16..23 (lo) / 24..31 (hi)
    }
    // --- stage 3: accumulate  out[l,i] += q[l, jb..jb+31] * K[j,i]
    U16 Aq;
    Aq.h[0] = *(const v8h*)(q0p + jb);
    Aq.h[1] = *(const v8h*)(q0p + jb + 16);
    c0 = __builtin_amdgcn_wmma_f32_16x16x32_f16(false, Aq.v, false, B2.v,
                                                (short)0, c0, false, false);
    Aq.h[0] = *(const v8h*)(q1p + jb);
    Aq.h[1] = *(const v8h*)(q1p + jb + 16);
    c1 = __builtin_amdgcn_wmma_f32_16x16x32_f16(false, Aq.v, false, B2.v,
                                                (short)0, c1, false, false);
  }

  // D layout: lane holds column i = i0+col; rows M=0..7 (lo) / 8..15 (hi).
#pragma unroll
  for (int r = 0; r < 8; ++r)                     // l = 0..15
    bl[(size_t)(half * 8 + r) * kN + i0 + col] = c0[r];
  if (lo) {
#pragma unroll
    for (int r = 0; r < 5; ++r)                   // l = 16..20
      bl[(size_t)(16 + r) * kN + i0 + col] = c1[r];
  } else {
    norm_b[i0 + col] = c1[7];                     // l = 31: the all-ones q row
  }
}

// --------------------------------------------------------------------- combine
// cur = unary + inv_s * (C@Ws) sp  + inv_b * (C@Wb) bl
// Weight indices are wave-uniform -> read straight from global so the compiler
// can scalarize them into s_load through the constant cache.
__global__ void k_combine(const float* __restrict__ unary,
                          const float* __restrict__ spf,
                          const float* __restrict__ bl,
                          const float* __restrict__ inv_ns,
                          const float* __restrict__ norm_b,
                          const float* __restrict__ cws,
                          const float* __restrict__ cwb,
                          float* __restrict__ cur) {
  int n = blockIdx.x * blockDim.x + threadIdx.x;
  if (n >= kN) return;
  float is = inv_ns[n];
  float ib = RCPF(norm_b[n]);
  float sp[kL], bv[kL];
#pragma unroll
  for (int m = 0; m < kL; ++m) {
    sp[m] = spf[(size_t)m * kN + n] * is;
    bv[m] = bl[(size_t)m * kN + n] * ib;
  }
  for (int l = 0; l < kL; ++l) {
    float a = unary[(size_t)l * kN + n];
#pragma unroll
    for (int m = 0; m < kL; ++m)
      a += cws[l * kL + m] * sp[m] + cwb[l * kL + m] * bv[m];
    cur[(size_t)l * kN + n] = a;
  }
}

// ----------------------------------------------------------------------- host
extern "C" void kernel_launch(void* const* d_in, const int* in_sizes, int n_in,
                              void* d_out, int out_size, void* d_ws, size_t ws_size,
                              hipStream_t stream) {
  const float* image  = (const float*)d_in[0];
  const float* logits = (const float*)d_in[1];
  const float* Wsp    = (const float*)d_in[2];
  const float* Wbl    = (const float*)d_in[3];
  const float* Cm     = (const float*)d_in[4];

  char* ws = (char*)d_ws;
  size_t off = 0;
  auto alloc = [&](size_t bytes) -> void* {
    void* p = ws + off;
    off = (off + bytes + 255) & ~(size_t)255;
    return p;
  };
  _Float16* featAp = (_Float16*)alloc((size_t)kN * 16 * 2);
  _Float16* featBp = (_Float16*)alloc((size_t)kN * 16 * 2);
  _Float16* qh     = (_Float16*)alloc((size_t)kLp * kN * 2);
  float* qf    = (float*)alloc((size_t)kL * kN * 4);
  float* t0    = (float*)alloc((size_t)kL * kN * 4);
  float* t1    = (float*)alloc((size_t)kL * kN * 4);
  float* bl    = (float*)alloc((size_t)kL * kN * 4);
  float* nrm_b = (float*)alloc((size_t)kN * 4);
  float* inv_s = (float*)alloc((size_t)kN * 4);
  float* cur   = (float*)alloc((size_t)kL * kN * 4);
  float* gS    = (float*)alloc(32 * 4);
  float* cws   = (float*)alloc(kL * kL * 4);
  float* cwb   = (float*)alloc(kL * kL * 4);
  (void)ws_size; (void)in_sizes; (void)n_in;

  const int TPB = 256;
  const int gN  = kN / TPB;           // 128
  const int gLN = (kL * kN) / TPB;    // 2688

  k_feat<<<gN, TPB, 0, stream>>>(image, featAp, featBp);
  k_misc<<<gN, TPB, 0, stream>>>(gS, inv_s, qh);
  k_foldw<<<2, 448, 0, stream>>>(Cm, Wsp, Wbl, cws, cwb);

  const float* curp = logits;
  for (int it = 0; it < 5; ++it) {
    k_softmax<<<gN, TPB, 0, stream>>>(curp, qf, qh);
    k_conv1d<1><<<gLN, TPB, 0, stream>>>(qf, t0, gS);      // x
    k_conv1d<32><<<gLN, TPB, 0, stream>>>(t0, t1, gS);     // y
    k_conv1d<1024><<<gLN, TPB, 0, stream>>>(t1, t0, gS);   // z
    k_bilateral<<<kN / 16 / 4, 128, 0, stream>>>(featAp, featBp, qh, bl, nrm_b);
    k_combine<<<gN, TPB, 0, stream>>>(logits, t0, bl, inv_s, nrm_b, cws, cwb, cur);
    curp = cur;
  }
  k_softmax<<<gN, TPB, 0, stream>>>(curp, (float*)d_out, nullptr);
  (void)out_size;
}
